// MyGraphSAGE_WithImage_87411174408886
// MI455X (gfx1250) — matrix operand order repaired and checked by reading, not compiled, not run on hardware
//
#include <hip/hip_runtime.h>
#include <hip/hip_bf16.h>

#define NN 50000

typedef __attribute__((ext_vector_type(16))) __bf16 v16bf;
typedef __attribute__((ext_vector_type(8)))  float  v8f;
typedef int v4i __attribute__((vector_size(16)));

__device__ __forceinline__ unsigned short f2bf(float f) {
  unsigned int u = __float_as_uint(f);
  u += 0x7fffu + ((u >> 16) & 1u);   // round-to-nearest-even
  return (unsigned short)(u >> 16);
}

// h0[n,0:128]=img[n], h0[n,128:256]=x[n]
__global__ void concat_kernel(const float* __restrict__ img, const float* __restrict__ x,
                              float* __restrict__ h0) {
  int tid = blockIdx.x * blockDim.x + threadIdx.x;   // over NN*256
  int n = tid >> 8, f = tid & 255;
  h0[tid] = (f < 128) ? img[n * 128 + f] : x[n * 128 + (f - 128)];
}

__global__ void degree_kernel(const long long* __restrict__ dst, float* __restrict__ deg, int E) {
  int e = blockIdx.x * blockDim.x + threadIdx.x;
  if (e < E) atomicAdd(deg + (int)dst[e], 1.0f);
}

__global__ void invdeg_kernel(const float* __restrict__ deg, float* __restrict__ inv) {
  int n = blockIdx.x * blockDim.x + threadIdx.x;
  if (n < NN) inv[n] = 1.0f / fmaxf(deg[n], 1.0f);
}

// one block (64 threads = 2 waves) per edge; F=256 -> 64 float4 chunks, fully coalesced
__global__ void scatter_kernel(const float* __restrict__ feat, const long long* __restrict__ src,
                               const long long* __restrict__ dst, float* __restrict__ agg) {
  int e = blockIdx.x;
  int c = threadIdx.x;   // 0..63
  int s = (int)src[e], d = (int)dst[e];
  const float4 v = ((const float4*)(feat + (size_t)s * 256))[c];
  float* o = agg + (size_t)d * 256 + c * 4;
  atomicAdd(o + 0, v.x);
  atomicAdd(o + 1, v.y);
  atomicAdd(o + 2, v.z);
  atomicAdd(o + 3, v.w);
}

// A[n, 0:256] = bf16(agg[n]*inv[n]); A[n, 256:512] = bf16(h[n])
__global__ void buildA_kernel(const float* __restrict__ agg, const float* __restrict__ inv,
                              const float* __restrict__ h, unsigned short* __restrict__ A) {
  int tid = blockIdx.x * blockDim.x + threadIdx.x;   // over NN*512
  int n = tid >> 9, k = tid & 511;
  float v = (k < 256) ? agg[(size_t)n * 256 + k] * inv[n]
                      : h[(size_t)n * 256 + (k - 256)];
  A[tid] = f2bf(v);
}

// Wc[o, 0:256] = Wl[o]; Wc[o, 256:512] = Wr[o]   (bf16, row-major [Fout,512])
__global__ void buildW_kernel(const float* __restrict__ Wl, const float* __restrict__ Wr,
                              unsigned short* __restrict__ Wc, int Fout) {
  int tid = blockIdx.x * blockDim.x + threadIdx.x;
  if (tid >= Fout * 512) return;
  int o = tid >> 9, k = tid & 511;
  Wc[tid] = f2bf((k < 256) ? Wl[o * 256 + k] : Wr[o * 256 + (k - 256)]);
}

// C[M,Nout] = relu(A[M,512] @ W[Nout,512]^T + bias).
// Block = (Nout/64) waves owning one 16-row stripe. The 16x512 bf16 A tile (16KB)
// is staged into LDS once via GLOBAL_LOAD_ASYNC_TO_LDS_B128 (ASYNCcnt), then each
// wave register-blocks 4 N-tiles: per K-step 2x ds_load_b128 (A) + 8x
// global_load_b128 (B, L2-resident) + 4x v_wmma_f32_16x16x32_bf16.
// Fragment layouts per CDNA5 ISA 7.12.2 (16-bit A: kb=half*8, elems 0..7->K+0..7,
// 8..15->K+16..23; 16-bit B: same K order as A (kb=half*8), column = lane&15).
__global__ void wmma_gemm_kernel(const unsigned short* __restrict__ A,
                                 const unsigned short* __restrict__ W,
                                 const float* __restrict__ bias,
                                 float* __restrict__ C, int Nout) {
  const int K = 512;
  __shared__ unsigned short ldsA[16 * 512];   // 16 KB

  const int tm = blockIdx.x * 16;

  // ---- stage A[tm:tm+16, 0:512] -> LDS (1024 x b128 chunks) ----
  for (int i = threadIdx.x; i < 1024; i += blockDim.x) {
    int r = i >> 6, c = (i & 63) * 8;
    const unsigned short* gp = A + (size_t)(tm + r) * K + c;
    unsigned short* lp = ldsA + r * 512 + c;
#if __has_builtin(__builtin_amdgcn_global_load_async_to_lds_b128)
    __builtin_amdgcn_global_load_async_to_lds_b128(
        (__attribute__((address_space(1))) v4i*)gp,
        (__attribute__((address_space(3))) v4i*)lp, 0, 0);
#else
    *(uint4*)lp = *(const uint4*)gp;
#endif
  }
#if __has_builtin(__builtin_amdgcn_global_load_async_to_lds_b128)
#if __has_builtin(__builtin_amdgcn_s_wait_asynccnt)
  __builtin_amdgcn_s_wait_asynccnt(0);
#else
  asm volatile("s_wait_asynccnt 0x0" ::: "memory");
#endif
#endif
  __syncthreads();

  const int lane = threadIdx.x & 31;
  const int wv   = threadIdx.x >> 5;        // 0..(Nout/64 - 1)
  const int half = lane >> 4;
  const int l15  = lane & 15;
  const int akb  = half * 8;
  const int bkb  = half * 8;

  const unsigned short* la = ldsA + l15 * 512;
  const int tn0 = wv * 64;
  const unsigned short* br[4];
#pragma unroll
  for (int nt = 0; nt < 4; ++nt)
    br[nt] = W + (size_t)(tn0 + nt * 16 + l15) * K;

  v8f acc[4] = {{}, {}, {}, {}};
#pragma unroll 4
  for (int k0 = 0; k0 < K; k0 += 32) {
    union { v16bf v; uint4 q[2]; } a, b[4];
    a.q[0] = *(const uint4*)(la + k0 + akb);          // K = k0+akb+0..7
    a.q[1] = *(const uint4*)(la + k0 + akb + 16);     // K = k0+akb+16..23
#pragma unroll
    for (int nt = 0; nt < 4; ++nt) {
      b[nt].q[0] = *(const uint4*)(br[nt] + k0 + bkb);
      b[nt].q[1] = *(const uint4*)(br[nt] + k0 + bkb + 16);
    }
#pragma unroll
    for (int nt = 0; nt < 4; ++nt)
      acc[nt] = __builtin_amdgcn_wmma_f32_16x16x32_bf16(
          false, a.v, false, b[nt].v, (short)0, acc[nt], false, false);
  }

#pragma unroll
  for (int nt = 0; nt < 4; ++nt) {
    int col = tn0 + nt * 16 + l15;
    float bv = bias[col];
#pragma unroll
    for (int r = 0; r < 8; ++r) {
      int row = tm + half * 8 + r;                    // C layout: VGPR r -> M = r + half*8
      C[(size_t)row * Nout + col] = fmaxf(acc[nt][r] + bv, 0.0f);
    }
  }
}

// One wave per node: 40-class FC + log_softmax, wave32 shuffle reductions.
__global__ void fc_lsm_kernel(const float* __restrict__ H,    // [NN,128]
                              const float* __restrict__ Wfc,  // [40,128]
                              const float* __restrict__ bfc,
                              float* __restrict__ out) {
  int gwave = (blockIdx.x * blockDim.x + threadIdx.x) >> 5;
  int lane = threadIdx.x & 31;
  if (gwave >= NN) return;
  const float* h = H + (size_t)gwave * 128;

  float lg0, lg1 = -3.0e38f;
  {
    const float* w = Wfc + lane * 128;
    float s = bfc[lane];
#pragma unroll 4
    for (int f = 0; f < 128; ++f) s += h[f] * w[f];
    lg0 = s;
  }
  bool has1 = (lane + 32) < 40;
  if (has1) {
    const float* w = Wfc + (lane + 32) * 128;
    float s = bfc[lane + 32];
#pragma unroll 4
    for (int f = 0; f < 128; ++f) s += h[f] * w[f];
    lg1 = s;
  }
  float m = fmaxf(lg0, lg1);
  for (int o = 16; o > 0; o >>= 1) m = fmaxf(m, __shfl_xor(m, o, 32));
  float se = __expf(lg0 - m) + (has1 ? __expf(lg1 - m) : 0.0f);
  for (int o = 16; o > 0; o >>= 1) se += __shfl_xor(se, o, 32);
  float lse = __logf(se) + m;
  out[(size_t)gwave * 40 + lane] = lg0 - lse;
  if (has1) out[(size_t)gwave * 40 + lane + 32] = lg1 - lse;
}

extern "C" void kernel_launch(void* const* d_in, const int* in_sizes, int n_in,
                              void* d_out, int out_size, void* d_ws, size_t ws_size,
                              hipStream_t stream) {
  (void)n_in; (void)out_size; (void)ws_size;
  const float*     x   = (const float*)d_in[0];
  const float*     img = (const float*)d_in[1];
  const long long* ei  = (const long long*)d_in[2];   // int64 edge_index [2,E]
  const float*     W1l = (const float*)d_in[3];
  const float*     b1l = (const float*)d_in[4];
  const float*     W1r = (const float*)d_in[5];
  const float*     W2l = (const float*)d_in[6];
  const float*     b2l = (const float*)d_in[7];
  const float*     W2r = (const float*)d_in[8];
  const float*     Wfc = (const float*)d_in[9];
  const float*     bfc = (const float*)d_in[10];
  const int E = in_sizes[2] / 2;
  const long long* src = ei;
  const long long* dst = ei + E;

  // workspace layout (~155 MB), h1 aliases h0, h2 aliases agg
  float* h0  = (float*)d_ws;                                         // NN*256 f32
  float* agg = h0 + (size_t)NN * 256;                                // NN*256 f32
  unsigned short* Abuf = (unsigned short*)(agg + (size_t)NN * 256);  // NN*512 bf16
  float* deg = (float*)(Abuf + (size_t)NN * 512);                    // NN f32
  float* inv = deg + NN;                                             // NN f32
  unsigned short* Wc1 = (unsigned short*)(inv + NN);                 // 256*512 bf16
  unsigned short* Wc2 = Wc1 + 256 * 512;                             // 128*512 bf16

  (void)hipMemsetAsync(deg, 0, NN * sizeof(float), stream);
  (void)hipMemsetAsync(agg, 0, (size_t)NN * 256 * sizeof(float), stream);

  concat_kernel<<<NN * 256 / 256, 256, 0, stream>>>(img, x, h0);
  degree_kernel<<<(E + 255) / 256, 256, 0, stream>>>(dst, deg, E);
  invdeg_kernel<<<(NN + 255) / 256, 256, 0, stream>>>(deg, inv);
  buildW_kernel<<<(256 * 512 + 255) / 256, 256, 0, stream>>>(W1l, W1r, Wc1, 256);
  buildW_kernel<<<(128 * 512 + 255) / 256, 256, 0, stream>>>(W2l, W2r, Wc2, 128);

  // ---- conv1: h1 = relu([agg(h0)*inv | h0] @ [W1l|W1r]^T + b1l) ----
  scatter_kernel<<<E, 64, 0, stream>>>(h0, src, dst, agg);
  buildA_kernel<<<NN * 512 / 256, 256, 0, stream>>>(agg, inv, h0, Abuf);
  wmma_gemm_kernel<<<NN / 16, 128, 0, stream>>>(Abuf, Wc1, b1l, h0 /* = h1 */, 256);

  // ---- conv2: h2 = relu([agg(h1)*inv | h1] @ [W2l|W2r]^T + b2l) ----
  (void)hipMemsetAsync(agg, 0, (size_t)NN * 256 * sizeof(float), stream);
  scatter_kernel<<<E, 64, 0, stream>>>(h0, src, dst, agg);
  buildA_kernel<<<NN * 512 / 256, 256, 0, stream>>>(agg, inv, h0, Abuf);
  float* h2 = agg;   // agg dead after buildA; reuse for h2 [NN,128]
  wmma_gemm_kernel<<<NN / 16, 64, 0, stream>>>(Abuf, Wc2, b2l, h2, 128);

  // ---- FC + log_softmax ----
  fc_lsm_kernel<<<(NN * 32 + 255) / 256, 256, 0, stream>>>(h2, Wfc, bfc, (float*)d_out);
}
